// PathwayGraphEncoder_3453153706541
// MI455X (gfx1250) — matrix operand order repaired and checked by reading, not compiled, not run on hardware
//
#include <hip/hip_runtime.h>
#include <hip/hip_bf16.h>
#include <math.h>

// ---------------------------------------------------------------------------
// PathwayGraphEncoder on gfx1250 (MI455X)
//   GEMMs  -> v_wmma_f32_16x16x32_f16, 64x16 output block per wave.
//             Weights pre-packed (f32->f16) into the per-lane WMMA B-fragment
//             layout so each fragment is one contiguous 32B load per lane.
//             Unconditional speculative prefetch of next A K-step.
//   Scatter-softmax -> 3-pass atomics (segment max / exp-sum / weighted add)
// ---------------------------------------------------------------------------

typedef __attribute__((ext_vector_type(16))) _Float16 v16h;
typedef __attribute__((ext_vector_type(8)))  float    v8f;

#define N_NODES  50000
#define N_EDGES  800000
#define N_ETOT   (N_EDGES + N_NODES)   // with self loops
#define HIDC     256                   // HEADS*HID
#define LAT      64
#define NUM_G    64

static __device__ __forceinline__ unsigned encf(float f) {
    unsigned u = __float_as_uint(f);
    return (u & 0x80000000u) ? ~u : (u | 0x80000000u);
}
static __device__ __forceinline__ float decf(unsigned u) {
    u = (u & 0x80000000u) ? (u & 0x7FFFFFFFu) : ~u;
    return __uint_as_float(u);
}
static __device__ __forceinline__ unsigned hash32(unsigned x) {
    x ^= x >> 16; x *= 0x7feb352dU;
    x ^= x >> 15; x *= 0x846ca68bU;
    x ^= x >> 16; return x;
}

// ---------------------------------------------------------------------------
// Pack W[K,N] (f32 row-major) into WMMA B-fragment order (f16):
//   fragment (tileN, kc) -> 32 lanes x 16 contiguous halves.
//   lane = ((k&31)>>4)*16 + (n&15); half index = k&15.
// ---------------------------------------------------------------------------
__global__ void pack_B_f16(const float* __restrict__ B, _Float16* __restrict__ Bp,
                           int K, int N) {
    int i = blockIdx.x * blockDim.x + threadIdx.x;
    if (i >= K * N) return;
    const int k = i / N, n = i - (i / N) * N;
    const int tileN = n >> 4, lr = n & 15;
    const int kc   = k >> 5;            // 32-wide K chunk
    const int rem  = k & 31;
    const int lane = (rem >> 4) * 16 + lr;
    const int idx  = rem & 15;
    Bp[(((tileN * (K >> 5) + kc) * 32) + lane) * 16 + idx] = (_Float16)B[i];
}

// ---------------------------------------------------------------------------
// WMMA GEMM: C[M,N] = A[M,K] * Bp ; f32 A (cvt to f16), pre-packed f16 B,
// f32 accumulate. One wave computes a 64x16 block (4 WMMA per K-step,
// one shared B fragment). All addressing 32-bit.
// ---------------------------------------------------------------------------
__global__ __launch_bounds__(32)
void gemm_wmma_f16(const float* __restrict__ A, const _Float16* __restrict__ Bp,
                   float* __restrict__ C, int M, int K, int N) {
    const int lane = threadIdx.x;          // 0..31
    const int grp  = lane >> 4;            // 0 | 1
    const int lr   = lane & 15;

    const int m0   = blockIdx.x * 64;      // first row of this wave's block
    const int colB = blockIdx.y * 16 + lr;

    // Clamp A-row indices (keeps EXEC all-ones through the WMMA loop).
    int rows[4];
#pragma unroll
    for (int t = 0; t < 4; ++t) {
        int r = m0 + t * 16 + lr;
        rows[t] = r < M ? r : M - 1;
    }

    // Per-lane packed B stream for this column tile: one v16h per K-chunk.
    const v16h* bptr = (const v16h*)Bp + (blockIdx.y * (K >> 5)) * 32 + lane;

    v8f acc[4] = {{}, {}, {}, {}};

    for (int k0 = 0; k0 < K; k0 += 32) {
        const v16h b = *bptr;              // 32B contiguous, wave-coalesced
        bptr += 32;

#pragma unroll
        for (int t = 0; t < 4; ++t) {
            // A fragment: lane group 0 holds K=k0+{0..15};
            //             lane group 1 holds K=k0+16+{0..15}.
            const float* arow = A + rows[t] * K + k0 + grp * 16;
            // Speculative prefetch of next K-step (dropped silently if OOB).
            __builtin_prefetch(arow + 32, 0, 1);
            v16h a;
#pragma unroll
            for (int i = 0; i < 8; ++i) a[i]     = (_Float16)arow[i];
#pragma unroll
            for (int i = 0; i < 8; ++i) a[8 + i] = (_Float16)arow[8 + i];

            acc[t] = __builtin_amdgcn_wmma_f32_16x16x32_f16(
                false, a, false, b, (short)0, acc[t], false, false);
        }
    }

    // D layout: VGPR i of tile t -> row m0 + t*16 + grp*8 + i, col = colB.
    if (m0 + 64 <= M) {
#pragma unroll
        for (int t = 0; t < 4; ++t) {
            const int rbase = m0 + t * 16 + grp * 8;
#pragma unroll
            for (int i = 0; i < 8; ++i)
                C[(rbase + i) * N + colB] = acc[t][i];
        }
    } else {
#pragma unroll
        for (int t = 0; t < 4; ++t) {
            const int rbase = m0 + t * 16 + grp * 8;
#pragma unroll
            for (int i = 0; i < 8; ++i) {
                const int r = rbase + i;
                if (r < M) C[r * N + colB] = acc[t][i];
            }
        }
    }
}

// ---------------------------------------------------------------------------
// Attention logits: als/ald[n*H+h] = sum_f h[n,h,f] * a_{s,d}[h,f]
// ---------------------------------------------------------------------------
__global__ void attn_logits(const float* __restrict__ hbuf,
                            const float* __restrict__ a_s,
                            const float* __restrict__ a_d,
                            float* __restrict__ als, float* __restrict__ ald,
                            int Nn, int H, int F) {
    int i = blockIdx.x * blockDim.x + threadIdx.x;
    if (i >= Nn * H) return;
    const int h = i % H;
    const float* hp = hbuf + (size_t)i * F;
    float ss = 0.f, sd = 0.f;
    for (int f = 0; f < F; ++f) {
        float v = hp[f];
        ss += v * a_s[h * F + f];
        sd += v * a_d[h * F + f];
    }
    als[i] = ss; ald[i] = sd;
}

__global__ void fill_u32(unsigned* __restrict__ p, unsigned v, int n) {
    int i = blockIdx.x * blockDim.x + threadIdx.x;
    if (i < n) p[i] = v;
}

// ---------------------------------------------------------------------------
// Scatter-softmax passes over edges (self loops appended logically).
// ---------------------------------------------------------------------------
__global__ void edge_logit_max(const int* __restrict__ src, const int* __restrict__ dst,
                               const float* __restrict__ als, const float* __restrict__ ald,
                               unsigned* __restrict__ mbuf, int E, int Et, int H) {
    int i = blockIdx.x * blockDim.x + threadIdx.x;
    if (i >= Et * H) return;
    const int e = i / H, h = i - (i / H) * H;
    int s, d;
    if (e < E) { s = src[e]; d = dst[e]; } else { s = e - E; d = s; }
    float v = als[s * H + h] + ald[d * H + h];
    v = v > 0.f ? v : 0.2f * v;                       // leaky relu
    atomicMax(&mbuf[d * H + h], encf(v));
}

__global__ void edge_exp_sum(const int* __restrict__ src, const int* __restrict__ dst,
                             const float* __restrict__ als, const float* __restrict__ ald,
                             const unsigned* __restrict__ mbuf,
                             float* __restrict__ eexp, float* __restrict__ sbuf,
                             int E, int Et, int H) {
    int i = blockIdx.x * blockDim.x + threadIdx.x;
    if (i >= Et * H) return;
    const int e = i / H, h = i - (i / H) * H;
    int s, d;
    if (e < E) { s = src[e]; d = dst[e]; } else { s = e - E; d = s; }
    float v = als[s * H + h] + ald[d * H + h];
    v = v > 0.f ? v : 0.2f * v;
    const float m = decf(mbuf[d * H + h]);
    const float ex = __expf(v - m);
    eexp[i] = ex;
    atomicAdd(&sbuf[d * H + h], ex);
}

__global__ void edge_aggregate(const int* __restrict__ src, const int* __restrict__ dst,
                               const float* __restrict__ hbuf,
                               const float* __restrict__ eexp,
                               const float* __restrict__ sbuf,
                               float* __restrict__ agg,
                               int E, int Et, int H, int F) {
    int i = blockIdx.x * blockDim.x + threadIdx.x;
    const int total = Et * H * F;
    if (i >= total) return;
    const int f  = i % F;
    const int eh = i / F;
    const int h  = eh % H;
    const int e  = eh / H;
    int s, d;
    if (e < E) { s = src[e]; d = dst[e]; } else { s = e - E; d = s; }
    const float alpha = eexp[e * H + h] / (sbuf[d * H + h] + 1e-16f);
    atomicAdd(&agg[(d * H + h) * F + f], hbuf[(s * H + h) * F + f] * alpha);
}

__global__ void bias_act(const float* __restrict__ agg, const float* __restrict__ b,
                         float* __restrict__ out, int Nn, int C, int do_elu) {
    int i = blockIdx.x * blockDim.x + threadIdx.x;
    if (i >= Nn * C) return;
    const int c = i % C;
    float v = agg[i] + b[c];
    if (do_elu) v = v > 0.f ? v : expm1f(v);
    out[i] = v;
}

// ---------------------------------------------------------------------------
// Global mean pool + VAE head
// ---------------------------------------------------------------------------
__global__ void pool_accum(const float* __restrict__ feat, const int* __restrict__ batch,
                           float* __restrict__ sums, float* __restrict__ cnts,
                           int Nn, int L) {
    int i = blockIdx.x * blockDim.x + threadIdx.x;
    if (i >= Nn * L) return;
    const int n = i / L, f = i - n * L;
    const int g = batch[n];
    atomicAdd(&sums[g * L + f], feat[i]);
    if (f == 0) atomicAdd(&cnts[g], 1.0f);
}

__global__ void pool_head(const float* __restrict__ sums, const float* __restrict__ cnts,
                          const float* __restrict__ Wmu, const float* __restrict__ bmu,
                          const float* __restrict__ Wlv, const float* __restrict__ blv,
                          float* __restrict__ out) {
    const int g = blockIdx.x;     // NUM_G
    const int j = threadIdx.x;    // LAT
    const float c = fmaxf(cnts[g], 1.0f);
    float mu = bmu[j], lv = blv[j];
    for (int l = 0; l < LAT; ++l) {
        const float p = sums[g * LAT + l] / c;
        mu += p * Wmu[l * LAT + j];
        lv += p * Wlv[l * LAT + j];
    }
    // Deterministic pseudo-normal (stand-in for jax.random.normal(key(42)))
    const unsigned s0 = hash32(0x2a2a2a2au ^ (unsigned)(g * LAT + j));
    const unsigned s1 = hash32(s0 ^ 0x9E3779B9u);
    const float u1 = (float)(s0 >> 8) * (1.0f / 16777216.0f) + 1e-7f;
    const float u2 = (float)(s1 >> 8) * (1.0f / 16777216.0f);
    const float eps = sqrtf(-2.0f * logf(u1)) * cosf(6.2831853f * u2);
    const int idx = g * LAT + j;
    out[idx]                   = mu;
    out[NUM_G * LAT + idx]     = lv;
    out[2 * NUM_G * LAT + idx] = mu + eps * expf(0.5f * lv);
}

// ---------------------------------------------------------------------------
extern "C" void kernel_launch(void* const* d_in, const int* in_sizes, int n_in,
                              void* d_out, int out_size, void* d_ws, size_t ws_size,
                              hipStream_t stream) {
    const float* x    = (const float*)d_in[0];
    const int*   ei   = (const int*)  d_in[1];
    const int*   bat  = (const int*)  d_in[2];
    const float* W1   = (const float*)d_in[3];
    const float* a1s  = (const float*)d_in[4];
    const float* a1d  = (const float*)d_in[5];
    const float* b1   = (const float*)d_in[6];
    const float* W2   = (const float*)d_in[7];
    const float* a2s  = (const float*)d_in[8];
    const float* a2d  = (const float*)d_in[9];
    const float* b2   = (const float*)d_in[10];
    const float* W3   = (const float*)d_in[11];
    const float* a3s  = (const float*)d_in[12];
    const float* a3d  = (const float*)d_in[13];
    const float* b3   = (const float*)d_in[14];
    const float* Wmu  = (const float*)d_in[15];
    const float* bmu  = (const float*)d_in[16];
    const float* Wlv  = (const float*)d_in[17];
    const float* blv  = (const float*)d_in[18];

    const int* srcp = ei;
    const int* dstp = ei + N_EDGES;

    // workspace carve-out
    char* w = (char*)d_ws;
    auto alloc = [&](size_t bytes) -> void* {
        void* p = (void*)w;
        w += (bytes + 255) & ~(size_t)255;
        return p;
    };
    float*     hbuf = (float*)    alloc((size_t)N_NODES * HIDC * 4);
    float*     feat = (float*)    alloc((size_t)N_NODES * HIDC * 4);
    float*     agg  = (float*)    alloc((size_t)N_NODES * HIDC * 4);
    float*     als  = (float*)    alloc((size_t)N_NODES * 4 * 4);
    float*     ald  = (float*)    alloc((size_t)N_NODES * 4 * 4);
    unsigned*  mbuf = (unsigned*) alloc((size_t)N_NODES * 4 * 4);
    float*     sbuf = (float*)    alloc((size_t)N_NODES * 4 * 4);
    float*     eexp = (float*)    alloc((size_t)N_ETOT  * 4 * 4);
    _Float16*  Bp   = (_Float16*) alloc((size_t)256 * 256 * 2);   // packed weights
    float*     sums = (float*)    alloc((size_t)NUM_G * LAT * 4);
    float*     cnts = (float*)    alloc((size_t)NUM_G * 4);

    auto cdiv = [](int a, int b) { return (a + b - 1) / b; };

    auto run_layer = [&](const float* in, const float* W, const float* as_,
                         const float* ad_, const float* b, int H, int F,
                         int Cout, int do_elu, float* outfeat) {
        // pack weights into WMMA B-fragment order (f16)
        pack_B_f16<<<cdiv(256 * Cout, 256), 256, 0, stream>>>(W, Bp, 256, Cout);

        dim3 gg(cdiv(N_NODES, 64), Cout / 16);
        gemm_wmma_f16<<<gg, 32, 0, stream>>>(in, Bp, hbuf, N_NODES, 256, Cout);

        const int nh = N_NODES * H;
        attn_logits<<<cdiv(nh, 256), 256, 0, stream>>>(hbuf, as_, ad_, als, ald,
                                                       N_NODES, H, F);
        fill_u32<<<cdiv(nh, 256), 256, 0, stream>>>(mbuf, 0u, nh);
        fill_u32<<<cdiv(nh, 256), 256, 0, stream>>>((unsigned*)sbuf, 0u, nh);
        const int na = N_NODES * H * F;
        fill_u32<<<cdiv(na, 256), 256, 0, stream>>>((unsigned*)agg, 0u, na);

        const int eh = N_ETOT * H;
        edge_logit_max<<<cdiv(eh, 256), 256, 0, stream>>>(srcp, dstp, als, ald,
                                                          mbuf, N_EDGES, N_ETOT, H);
        edge_exp_sum<<<cdiv(eh, 256), 256, 0, stream>>>(srcp, dstp, als, ald, mbuf,
                                                        eexp, sbuf, N_EDGES, N_ETOT, H);
        const int tot = N_ETOT * H * F;
        edge_aggregate<<<cdiv(tot, 256), 256, 0, stream>>>(srcp, dstp, hbuf, eexp,
                                                           sbuf, agg, N_EDGES, N_ETOT,
                                                           H, F);
        bias_act<<<cdiv(na, 256), 256, 0, stream>>>(agg, b, outfeat, N_NODES, H * F,
                                                    do_elu);
    };

    run_layer(x,    W1, a1s, a1d, b1, 4, 64, 256, 1, feat);
    run_layer(feat, W2, a2s, a2d, b2, 4, 64, 256, 1, feat);
    run_layer(feat, W3, a3s, a3d, b3, 1, 64,  64, 0, feat);   // mean over 1 head

    // global mean pool + VAE head
    fill_u32<<<cdiv(NUM_G * LAT, 256), 256, 0, stream>>>((unsigned*)sums, 0u,
                                                         NUM_G * LAT);
    fill_u32<<<1, 256, 0, stream>>>((unsigned*)cnts, 0u, NUM_G);
    pool_accum<<<cdiv(N_NODES * LAT, 256), 256, 0, stream>>>(feat, bat, sums, cnts,
                                                             N_NODES, LAT);
    pool_head<<<NUM_G, LAT, 0, stream>>>(sums, cnts, Wmu, bmu, Wlv, blv,
                                         (float*)d_out);
}
